// cross_att_47021301956873
// MI455X (gfx1250) — matrix operand order repaired and checked by reading, not compiled, not run on hardware
//
#include <hip/hip_runtime.h>
#include <hip/hip_bf16.h>
#include <math.h>

// ---------------------------------------------------------------------------
// Channel cross-attention + residual + LayerNorm for MI455X (gfx1250, wave32)
// One workgroup (8 waves / 256 threads) per (b, n) slab; bf16 WMMA matmuls.
// All LDS WMMA fragments are two 16B per-lane runs (p[0..7], p[16..23]) -> 2x ds_load_b128.
// ---------------------------------------------------------------------------

typedef __attribute__((ext_vector_type(16))) __bf16 v16bf;
typedef __attribute__((ext_vector_type(8)))  float  v8f;

#define B_  32
#define D_  128
#define N_  64
#define L_  96
#define H_  8
#define NL_ (N_ * L_)   // 6144, stride between channel rows in x/x2

// LDS layout. Row strides chosen so rows are 16B-aligned and bank-friendly.
constexpr int RSX = 104;  // row stride (bf16) for [128 x 96]  row-major tiles (208 B)
constexpr int RST = 136;  // row stride (bf16) for [96 x 128] transposed tiles (272 B)
constexpr int RSA = 136;  // row stride (bf16) for [128 x 128] attn tile       (272 B)
constexpr int SZ_T = L_ * RST * 2;   // 26112 B  (transposed 96x128 tile)
constexpr int SZ_R = D_ * RSX * 2;   // 26624 B  (row-major 128x96 tile)
constexpr int OFF_X2T = 0;                       // x2 transposed [l][d]
constexpr int OFF_XT  = OFF_X2T + SZ_T;          // x  transposed [l][d]
constexpr int OFF_QB  = OFF_XT  + SZ_T;          // Q [e][l]
constexpr int OFF_KB  = OFF_QB  + SZ_R;          // K [f][l]
constexpr int OFF_VT  = OFF_KB  + SZ_R;          // V transposed [l][f]
constexpr int LDS_BYTES = OFF_VT + SZ_T;         // 131584 B (< 320 KB WGP LDS)
constexpr int OFF_AB  = 0;  // attn [e][f], aliases X2T+XT (free after stage 1)

union Frag16 { uint4 q[2]; v16bf v; };
union PairBf { __bf16 h[2]; unsigned int u; };

__device__ __forceinline__ __bf16 to_bf16(float f) { return (__bf16)f; }

__device__ __forceinline__ v8f zero_v8f() {
  v8f z;
#pragma unroll
  for (int r = 0; r < 8; ++r) z[r] = 0.0f;
  return z;
}

// 16 consecutive bf16 at p (32 B, 16B-aligned): not used by the fragment loads.
__device__ __forceinline__ v16bf load_frag_contig(const __bf16* p) {
  Frag16 f;
  f.q[0] = *(const uint4*)(p);
  f.q[1] = *(const uint4*)(p + 8);
  return f.v;
}

// A-fragment pattern: elements p[0..7] and p[16..23] (ISA 7.12.2, 16-bit A).
__device__ __forceinline__ v16bf load_frag_a_pat(const __bf16* p) {
  Frag16 f;
  f.q[0] = *(const uint4*)(p);
  f.q[1] = *(const uint4*)(p + 16);
  return f.v;
}

// A-fragment (16x32, MxK) from an LDS row-major tile.
__device__ __forceinline__ v16bf load_a_frag(const __bf16* base, int stride,
                                             int r0, int c0, int lane) {
  const int hf = lane >> 4, lr = lane & 15;
  return load_frag_a_pat(base + (r0 + lr) * stride + c0 + 8 * hf);
}

// B-fragment (32x16, KxN) where source holds S[n][k] (same k pattern as A).
__device__ __forceinline__ v16bf load_b_fragT(const __bf16* base, int stride,
                                              int k0, int n0, int lane) {
  const int hf = lane >> 4, ln = lane & 15;
  return load_frag_a_pat(base + (n0 + ln) * stride + k0 + 8 * hf);
}

// A-fragment gathered from global fp32 weight matrix (row-major DxD),
// converted to bf16 (L2-resident; compiler emits b128 loads + cvt_pk).
__device__ __forceinline__ v16bf load_a_frag_w(const float* __restrict__ W,
                                               int r0, int c0, int lane) {
  const int hf = lane >> 4, lr = lane & 15;
  const float* row = W + (r0 + lr) * D_ + c0 + 8 * hf;
  const float4 a0 = *(const float4*)(row);
  const float4 a1 = *(const float4*)(row + 4);
  const float4 a2 = *(const float4*)(row + 16);
  const float4 a3 = *(const float4*)(row + 20);
  v16bf f;
  f[0] = to_bf16(a0.x); f[1] = to_bf16(a0.y); f[2]  = to_bf16(a0.z); f[3]  = to_bf16(a0.w);
  f[4] = to_bf16(a1.x); f[5] = to_bf16(a1.y); f[6]  = to_bf16(a1.z); f[7]  = to_bf16(a1.w);
  f[8] = to_bf16(a2.x); f[9] = to_bf16(a2.y); f[10] = to_bf16(a2.z); f[11] = to_bf16(a2.w);
  f[12] = to_bf16(a3.x); f[13] = to_bf16(a3.y); f[14] = to_bf16(a3.z); f[15] = to_bf16(a3.w);
  return f;
}

// dst[e][l] = sum_d W[e][d]*srcT[l][d] + bias[e]; row-major output (Q, K).
__device__ __forceinline__ void project_rm(const __bf16* srcT, const float* __restrict__ W,
                                           const float* __restrict__ bias,
                                           __bf16* dst, int m0, int lane) {
  v8f acc[6];
#pragma unroll
  for (int j = 0; j < 6; ++j) acc[j] = zero_v8f();
#pragma unroll
  for (int kt = 0; kt < 4; ++kt) {
    const v16bf a = load_a_frag_w(W, m0, kt * 32, lane);
#pragma unroll
    for (int j = 0; j < 6; ++j) {
      const v16bf bmat = load_b_fragT(srcT, RST, kt * 32, j * 16, lane);
      acc[j] = __builtin_amdgcn_wmma_f32_16x16x32_bf16(
          false, a, false, bmat, (short)0, acc[j], false, false);
    }
  }
  const int hf = lane >> 4, ln = lane & 15;
#pragma unroll
  for (int r = 0; r < 8; ++r) {
    const int   e  = m0 + r + 8 * hf;
    const float bb = bias[e];
#pragma unroll
    for (int j = 0; j < 6; ++j)
      dst[e * RSX + j * 16 + ln] = to_bf16(acc[j][r] + bb);
  }
}

// Same math, but output stored transposed: dstT[l][e] (V). Paired b32 stores.
__device__ __forceinline__ void project_tr(const __bf16* srcT, const float* __restrict__ W,
                                           const float* __restrict__ bias,
                                           __bf16* dstT, int m0, int lane) {
  v8f acc[6];
#pragma unroll
  for (int j = 0; j < 6; ++j) acc[j] = zero_v8f();
#pragma unroll
  for (int kt = 0; kt < 4; ++kt) {
    const v16bf a = load_a_frag_w(W, m0, kt * 32, lane);
#pragma unroll
    for (int j = 0; j < 6; ++j) {
      const v16bf bmat = load_b_fragT(srcT, RST, kt * 32, j * 16, lane);
      acc[j] = __builtin_amdgcn_wmma_f32_16x16x32_bf16(
          false, a, false, bmat, (short)0, acc[j], false, false);
    }
  }
  const int hf = lane >> 4, ln = lane & 15;
#pragma unroll
  for (int r = 0; r < 8; r += 2) {
    const int e = m0 + r + 8 * hf;           // even
    const float b0 = bias[e], b1 = bias[e + 1];
#pragma unroll
    for (int j = 0; j < 6; ++j) {
      PairBf p;
      p.h[0] = to_bf16(acc[j][r] + b0);
      p.h[1] = to_bf16(acc[j][r + 1] + b1);
      *(unsigned int*)(dstT + (j * 16 + ln) * RST + e) = p.u;
    }
  }
}

__global__ __launch_bounds__(256)
void cross_att_47021301956873_kernel(const float* __restrict__ x,
                                     const float* __restrict__ x2,
                                     const float* __restrict__ Wq, const float* __restrict__ bq,
                                     const float* __restrict__ Wk, const float* __restrict__ bk,
                                     const float* __restrict__ Wv, const float* __restrict__ bv,
                                     const float* __restrict__ Wo, const float* __restrict__ bo,
                                     const float* __restrict__ gamma,
                                     const float* __restrict__ beta,
                                     float* __restrict__ out) {
  __shared__ __align__(16) char smem[LDS_BYTES];
  __bf16* X2T = (__bf16*)(smem + OFF_X2T);
  __bf16* XT  = (__bf16*)(smem + OFF_XT);
  __bf16* QB  = (__bf16*)(smem + OFF_QB);
  __bf16* KB  = (__bf16*)(smem + OFF_KB);
  __bf16* VT  = (__bf16*)(smem + OFF_VT);
  __bf16* AB  = (__bf16*)(smem + OFF_AB);

  const int tid  = threadIdx.x;
  const int lane = tid & 31;
  const int w    = tid >> 5;          // wave 0..7
  const int bid  = blockIdx.x;        // 0..2047
  const int b    = bid >> 6;
  const int n    = bid & 63;
  const int hf   = lane >> 4;
  const int ln   = lane & 15;
  const int m0   = w * 16;

  const float* xs  = x  + (size_t)b * D_ * NL_ + (size_t)n * L_;
  const float* x2s = x2 + (size_t)b * D_ * NL_ + (size_t)n * L_;

  // ---- Stage 0: global fp32 -> LDS bf16, stored TRANSPOSED [l][d] ---------
  for (int i = tid; i < D_ * (L_ / 4); i += 256) {
    const int d = i / (L_ / 4);
    const int c = (i % (L_ / 4)) * 4;
    const float4 vx = *(const float4*)(xs  + (size_t)d * NL_ + c);
    const float4 vy = *(const float4*)(x2s + (size_t)d * NL_ + c);
    XT[(c + 0) * RST + d] = to_bf16(vx.x);
    XT[(c + 1) * RST + d] = to_bf16(vx.y);
    XT[(c + 2) * RST + d] = to_bf16(vx.z);
    XT[(c + 3) * RST + d] = to_bf16(vx.w);
    X2T[(c + 0) * RST + d] = to_bf16(vy.x);
    X2T[(c + 1) * RST + d] = to_bf16(vy.y);
    X2T[(c + 2) * RST + d] = to_bf16(vy.z);
    X2T[(c + 3) * RST + d] = to_bf16(vy.w);
  }
  __syncthreads();

  // ---- Stage 1: Q/K/V projections (bf16 WMMA, fp32 accumulate) ------------
  project_rm(X2T, Wq, bq, QB, m0, lane);   // Q [e][l]
  project_rm(XT,  Wk, bk, KB, m0, lane);   // K [f][l]
  project_tr(XT,  Wv, bv, VT, m0, lane);   // V -> VT [l][f]
  __syncthreads();

  // ---- Stage 2: scores = Q K^T / 4, softmax in registers, attn -> LDS -----
  {
    v8f s[8];
#pragma unroll
    for (int j = 0; j < 8; ++j) s[j] = zero_v8f();
#pragma unroll
    for (int lt = 0; lt < 3; ++lt) {
      const v16bf a = load_a_frag(QB, RSX, m0, lt * 32, lane);
#pragma unroll
      for (int j = 0; j < 8; ++j) {
        // B[k=l][n=f] = K[f][l]: contiguous in KB's row-major [f][l] layout.
        const v16bf bmat = load_b_fragT(KB, RSX, lt * 32, j * 16, lane);
        s[j] = __builtin_amdgcn_wmma_f32_16x16x32_bf16(
            false, a, false, bmat, (short)0, s[j], false, false);
      }
    }
    // Row softmax: row e = m0 + r + 8*hf lives in one 16-lane half, 8 tiles.
#pragma unroll
    for (int r = 0; r < 8; ++r) {
      float mx = -3.4e38f;
#pragma unroll
      for (int j = 0; j < 8; ++j) { s[j][r] *= 0.25f; mx = fmaxf(mx, s[j][r]); }
#pragma unroll
      for (int msk = 1; msk < 16; msk <<= 1) mx = fmaxf(mx, __shfl_xor(mx, msk, 32));
      float sum = 0.0f;
#pragma unroll
      for (int j = 0; j < 8; ++j) { const float e = __expf(s[j][r] - mx); s[j][r] = e; sum += e; }
#pragma unroll
      for (int msk = 1; msk < 16; msk <<= 1) sum += __shfl_xor(sum, msk, 32);
      const float inv = 1.0f / sum;
      const int   e   = m0 + r + 8 * hf;
#pragma unroll
      for (int j = 0; j < 8; ++j)
        AB[e * RSA + j * 16 + ln] = to_bf16(s[j][r] * inv);
    }
  }
  __syncthreads();

  // ---- Stage 3: line = attn x V; residual (fp32 x, L2-hot) + LayerNorm ----
  {
    v8f acc[6];
#pragma unroll
    for (int j = 0; j < 6; ++j) acc[j] = zero_v8f();
#pragma unroll
    for (int kt = 0; kt < 4; ++kt) {
      const v16bf a = load_a_frag(AB, RSA, m0, kt * 32, lane);
#pragma unroll
      for (int j = 0; j < 6; ++j) {
        // B[k=f][n=l] = V[f][l] = VT[l][f]: contiguous in VT.
        const v16bf bmat = load_b_fragT(VT, RST, kt * 32, j * 16, lane);
        acc[j] = __builtin_amdgcn_wmma_f32_16x16x32_bf16(
            false, a, false, bmat, (short)0, acc[j], false, false);
      }
    }
    float sWo = 0.0f;
#pragma unroll
    for (int q = 0; q < H_; ++q) sWo += Wo[q];
    const float bo0 = bo[0];
    float* outs = out + (size_t)(b * N_ + n) * D_ * L_;

#pragma unroll
    for (int r = 0; r < 8; ++r) {
      const int    e    = m0 + r + 8 * hf;
      const float* xrow = xs + (size_t)e * NL_;   // residual row (L2-hot)
      float hv[6];
      float sum = 0.0f, sq = 0.0f;
#pragma unroll
      for (int j = 0; j < 6; ++j) {
        const float val = xrow[j * 16 + ln] + acc[j][r] * sWo + bo0;
        hv[j] = val;
        sum += val; sq += val * val;
      }
#pragma unroll
      for (int msk = 1; msk < 16; msk <<= 1) {
        sum += __shfl_xor(sum, msk, 32);
        sq  += __shfl_xor(sq,  msk, 32);
      }
      const float mu  = sum * (1.0f / L_);
      const float var = sq * (1.0f / L_) - mu * mu;
      const float rs  = rsqrtf(var + 1e-5f);
#pragma unroll
      for (int j = 0; j < 6; ++j) {
        const int l = j * 16 + ln;
        outs[(size_t)e * L_ + l] = (hv[j] - mu) * rs * gamma[l] + beta[l];
      }
    }
  }
}

extern "C" void kernel_launch(void* const* d_in, const int* in_sizes, int n_in,
                              void* d_out, int out_size, void* d_ws, size_t ws_size,
                              hipStream_t stream) {
  (void)in_sizes; (void)n_in; (void)out_size; (void)d_ws; (void)ws_size;
  const float* x     = (const float*)d_in[0];
  const float* x2    = (const float*)d_in[1];
  const float* Wq    = (const float*)d_in[2];
  const float* bq    = (const float*)d_in[3];
  const float* Wk    = (const float*)d_in[4];
  const float* bk    = (const float*)d_in[5];
  const float* Wv    = (const float*)d_in[6];
  const float* bv    = (const float*)d_in[7];
  const float* Wo    = (const float*)d_in[8];
  const float* bo    = (const float*)d_in[9];
  const float* gamma = (const float*)d_in[10];
  const float* beta  = (const float*)d_in[11];
  float* out = (float*)d_out;

  dim3 grid(B_ * N_), block(256);
  cross_att_47021301956873_kernel<<<grid, block, 0, stream>>>(
      x, x2, Wq, bq, Wk, bk, Wv, bv, Wo, bo, gamma, beta, out);
}
